// NoCrossAttention_49048526520632
// MI455X (gfx1250) — matrix operand order repaired and checked by reading, not compiled, not run on hardware
//
#include <hip/hip_runtime.h>
#include <hip/hip_bf16.h>

typedef __attribute__((ext_vector_type(16))) __bf16 v16bf;
typedef __attribute__((ext_vector_type(8)))  float  v8f;

#define HDIM   256
#define LAYERS 5
#define BGRAPH 1024
#define NLIG   65536
#define NPROT  131072
#define ELIG   262144
#define EPROT  524288
#define EPS_BN 1e-5f

// ---------- helpers ----------
__device__ __forceinline__ unsigned short f2bf(float f) {
  unsigned u = __float_as_uint(f);
  u += 0x7fffu + ((u >> 16) & 1u);          // round-to-nearest-even
  return (unsigned short)(u >> 16);
}

union ABits { v16bf v; unsigned short u[16]; uint4 q[2]; };

// ---------- embedding: h = x @ W + b (K = 26 or 20) ----------
__global__ __launch_bounds__(256)
void embed_kernel(const float* __restrict__ x, const float* __restrict__ W,
                  const float* __restrict__ b, float* __restrict__ h, int K) {
  __shared__ float xs[32];
  const int n = blockIdx.x;
  const int c = threadIdx.x;
  if (c < K) xs[c] = x[(size_t)n * K + c];
  __syncthreads();
  float acc = b[c];
  for (int k = 0; k < K; ++k) acc += xs[k] * W[(size_t)k * HDIM + c];
  h[(size_t)n * HDIM + c] = acc;
}

// ---------- broadcast virtual0 into v [B,H] ----------
__global__ __launch_bounds__(256)
void bcast_init(const float* __restrict__ v0, float* __restrict__ v) {
  const int gid = blockIdx.x * 256 + threadIdx.x;
  v[gid] = v0[gid & (HDIM - 1)];
}

// ---------- v[b] += sum over contiguous segment of h ----------
__global__ __launch_bounds__(256)
void pool_add(const float* __restrict__ h, float* __restrict__ v, int seg) {
  const int b = blockIdx.x, c = threadIdx.x;
  const float* p = h + (size_t)b * seg * HDIM + c;
  float s = 0.f;
  for (int r = 0; r < seg; ++r) s += p[(size_t)r * HDIM];
  v[b * HDIM + c] += s;
}

// ---------- mean-pool a segment into combined [B, 2H] ----------
__global__ __launch_bounds__(256)
void pool_mean(const float* __restrict__ h, float* __restrict__ comb, int seg, int off) {
  const int b = blockIdx.x, c = threadIdx.x;
  const float* p = h + (size_t)b * seg * HDIM + c;
  float s = 0.f;
  for (int r = 0; r < seg; ++r) s += p[(size_t)r * HDIM];
  comb[(size_t)b * (2 * HDIM) + off + c] = s / (float)seg;
}

// ---------- pack fp32 W[K,256] into bf16 WMMA B-fragment layout ----------
// tile (kt,nt): 32x16 block at k0=kt*32, n0=nt*16.  Lane (half,lr):
//   col = nt*16+lr,  k(j) = kt*32 + half*8 + (j < 8 ? j : j + 8)   (j = 0..15, 2 bf16 per VGPR)
__global__ __launch_bounds__(32)
void pack_w(const float* __restrict__ W, unsigned short* __restrict__ Wp) {
  const int tile = blockIdx.x;          // kt*(H/16) + nt
  const int lane = threadIdx.x;
  const int kt = tile >> 4, nt = tile & 15;
  const int half = lane >> 4, lr = lane & 15;
  const int col = nt * 16 + lr;
  unsigned short* o = Wp + ((size_t)tile * 32 + lane) * 16;
#pragma unroll
  for (int j = 0; j < 16; ++j) {
    const int k = kt * 32 + half * 8 + (j < 8 ? j : j + 8);
    o[j] = f2bf(W[(size_t)k * HDIM + col]);
  }
}

// ---------- main WMMA GEMM: C[M,256] = op(A [+A2]) @ W + bias [+Cres] [relu] ----------
// block = 256 threads = 8 waves; block tile 32 rows x 256 cols; wave tile 16x64.
__global__ __launch_bounds__(256)
void gemm256_wmma(const float* __restrict__ A, const float* __restrict__ A2, int lda,
                  const unsigned short* __restrict__ Wp, const float* __restrict__ bias,
                  const float* __restrict__ Cres, float* __restrict__ C,
                  int K, int do_relu) {
  const int tid  = threadIdx.x;
  const int wave = tid >> 5;
  const int lane = tid & 31;
  const int half = lane >> 4;
  const int lr   = lane & 15;
  const int m0 = blockIdx.x * 32 + (wave >> 2) * 16;
  const int n0 = (wave & 3) * 64;
  const int row = m0 + lr;

  const float* arow  = A + (size_t)row * lda;
  const float* arow2 = A2 ? (A2 + (size_t)row * lda) : nullptr;

  v8f acc[4] = {};

  for (int k0 = 0; k0 < K; k0 += 32) {
    const int kb = k0 + half * 8;
    const float4* p0 = (const float4*)(arow + kb);
    const float4* p1 = (const float4*)(arow + kb + 16);
    float4 a0 = p0[0], a1 = p0[1], a2 = p1[0], a3 = p1[1];
    if (arow2) {
      const float4* q0 = (const float4*)(arow2 + kb);
      const float4* q1 = (const float4*)(arow2 + kb + 16);
      float4 b0 = q0[0], b1 = q0[1], b2 = q1[0], b3 = q1[1];
      a0.x += b0.x; a0.y += b0.y; a0.z += b0.z; a0.w += b0.w;
      a1.x += b1.x; a1.y += b1.y; a1.z += b1.z; a1.w += b1.w;
      a2.x += b2.x; a2.y += b2.y; a2.z += b2.z; a2.w += b2.w;
      a3.x += b3.x; a3.y += b3.y; a3.z += b3.z; a3.w += b3.w;
    }
    ABits af;
    af.u[0]  = f2bf(a0.x); af.u[1]  = f2bf(a0.y); af.u[2]  = f2bf(a0.z); af.u[3]  = f2bf(a0.w);
    af.u[4]  = f2bf(a1.x); af.u[5]  = f2bf(a1.y); af.u[6]  = f2bf(a1.z); af.u[7]  = f2bf(a1.w);
    af.u[8]  = f2bf(a2.x); af.u[9]  = f2bf(a2.y); af.u[10] = f2bf(a2.z); af.u[11] = f2bf(a2.w);
    af.u[12] = f2bf(a3.x); af.u[13] = f2bf(a3.y); af.u[14] = f2bf(a3.z); af.u[15] = f2bf(a3.w);

    const int kt = k0 >> 5;
    const size_t tbase = ((size_t)(kt * (HDIM / 16) + (n0 >> 4)) * 32 + lane) * 16;
#pragma unroll
    for (int t = 0; t < 4; ++t) {
      ABits bf;
      const uint4* wq = (const uint4*)(Wp + tbase + (size_t)t * 512);
      bf.q[0] = wq[0];
      bf.q[1] = wq[1];
      acc[t] = __builtin_amdgcn_wmma_f32_16x16x32_bf16(false, af.v, false, bf.v,
                                                       (short)0, acc[t], false, false);
    }
  }

#pragma unroll
  for (int t = 0; t < 4; ++t) {
    const int col = n0 + t * 16 + lr;
    const float bb = bias[col];
#pragma unroll
    for (int g = 0; g < 8; ++g) {
      const int r = m0 + half * 8 + g;
      float val = acc[t][g] + bb;
      if (Cres)    val += Cres[(size_t)r * HDIM + col];
      if (do_relu) val = fmaxf(val, 0.f);
      C[(size_t)r * HDIM + col] = val;
    }
  }
}

// ---------- column stats over [B,H] (biased var) ----------
__global__ __launch_bounds__(256)
void colstats(const float* __restrict__ X, float* __restrict__ mean, float* __restrict__ var) {
  const int c = blockIdx.x, t = threadIdx.x;
  float s = 0.f, sq = 0.f;
  for (int r = t; r < BGRAPH; r += 256) {
    const float x = X[(size_t)r * HDIM + c];
    s += x; sq += x * x;
  }
  __shared__ float ss[256], sqq[256];
  ss[t] = s; sqq[t] = sq; __syncthreads();
  for (int o = 128; o > 0; o >>= 1) {
    if (t < o) { ss[t] += ss[t + o]; sqq[t] += sqq[t + o]; }
    __syncthreads();
  }
  if (t == 0) {
    const float m = ss[0] / (float)BGRAPH;
    mean[c] = m;
    var[c]  = sqq[0] / (float)BGRAPH - m * m;
  }
}

// ---------- BN + ReLU apply ----------
__global__ __launch_bounds__(256)
void bn_relu_apply(const float* __restrict__ X, const float* __restrict__ mean,
                   const float* __restrict__ var, const float* __restrict__ gamma,
                   const float* __restrict__ beta, float* __restrict__ V) {
  const int gid = blockIdx.x * 256 + threadIdx.x;
  const int c = gid & (HDIM - 1);
  const float y = (X[gid] - mean[c]) * rsqrtf(var[c] + EPS_BN) * gamma[c] + beta[c];
  V[gid] = fmaxf(y, 0.f);
}

// ---------- h[n] += v[n >> shift] ----------
__global__ __launch_bounds__(256)
void vadd_bcast(float* __restrict__ h, const float* __restrict__ v, int shift) {
  const size_t gid = (size_t)blockIdx.x * 256 + threadIdx.x;
  const int c = (int)(gid & (HDIM - 1));
  const size_t n = gid >> 8;
  h[gid] += v[((n >> shift) << 8) + c];
}

// ---------- zero ----------
__global__ __launch_bounds__(256)
void zero_kernel(float* __restrict__ p, size_t n) {
  const size_t g = (size_t)blockIdx.x * 256 + threadIdx.x;
  if (g < n) p[g] = 0.f;
}

// ---------- edge scatter-add: one wave per edge, coalesced atomics ----------
__global__ __launch_bounds__(256)
void edge_scatter(const float* __restrict__ h, const int* __restrict__ src,
                  const int* __restrict__ dst, float* __restrict__ agg, int E) {
  const size_t gid = (size_t)blockIdx.x * 256 + threadIdx.x;
  const int e = (int)(gid >> 5);
  if (e >= E) return;
  const int c0 = (int)(gid & 31);
  const size_t so = (size_t)src[e] * HDIM;
  const size_t dof = (size_t)dst[e] * HDIM;
#pragma unroll
  for (int i = 0; i < 8; ++i) {
    const int col = c0 + i * 32;
    atomicAdd(agg + dof + col, h[so + col]);
  }
}

// ---------- final: out[b] = relu_h1[b,:] . W2 + b2 ----------
__global__ __launch_bounds__(256)
void final_dot(const float* __restrict__ h1, const float* __restrict__ W2,
               const float* __restrict__ b2, float* __restrict__ out) {
  const int row = blockIdx.x * 8 + (threadIdx.x >> 5);
  const int lane = threadIdx.x & 31;
  float s = 0.f;
  for (int k = lane; k < HDIM; k += 32) s += h1[(size_t)row * HDIM + k] * W2[k];
  for (int o = 16; o > 0; o >>= 1) s += __shfl_down(s, o, 32);
  if (lane == 0) out[row] = s + b2[0];
}

// ============================ host ============================
extern "C" void kernel_launch(void* const* d_in, const int* in_sizes, int n_in,
                              void* d_out, int out_size, void* d_ws, size_t ws_size,
                              hipStream_t stream) {
  const float* lig_x        = (const float*)d_in[0];
  const float* prot_x       = (const float*)d_in[1];
  const float* lig_emb_W    = (const float*)d_in[2];
  const float* lig_emb_b    = (const float*)d_in[3];
  const float* prot_emb_W   = (const float*)d_in[4];
  const float* prot_emb_b   = (const float*)d_in[5];
  const float* lig_v0       = (const float*)d_in[6];
  const float* prot_v0      = (const float*)d_in[7];
  const float* lig_cW1      = (const float*)d_in[8];
  const float* lig_cb1      = (const float*)d_in[9];
  const float* lig_cW2      = (const float*)d_in[10];
  const float* lig_cb2      = (const float*)d_in[11];
  const float* lig_vW       = (const float*)d_in[12];
  const float* lig_vb       = (const float*)d_in[13];
  const float* lig_vg       = (const float*)d_in[14];
  const float* lig_vbeta    = (const float*)d_in[15];
  const float* prot_cW1     = (const float*)d_in[16];
  const float* prot_cb1     = (const float*)d_in[17];
  const float* prot_cW2     = (const float*)d_in[18];
  const float* prot_cb2     = (const float*)d_in[19];
  const float* prot_vW      = (const float*)d_in[20];
  const float* prot_vb      = (const float*)d_in[21];
  const float* prot_vg      = (const float*)d_in[22];
  const float* prot_vbeta   = (const float*)d_in[23];
  const float* pred_W1      = (const float*)d_in[24];
  const float* pred_b1      = (const float*)d_in[25];
  const float* pred_W2      = (const float*)d_in[26];
  const float* pred_b2      = (const float*)d_in[27];
  const int*   lig_edge     = (const int*)d_in[30];
  const int*   prot_edge    = (const int*)d_in[31];

  float* ws = (float*)d_ws;
  size_t off = 0;
  float* h_l  = ws + off; off += (size_t)NLIG  * HDIM;
  float* h_p  = ws + off; off += (size_t)NPROT * HDIM;
  float* agg  = ws + off; off += (size_t)NPROT * HDIM;   // shared scratch, sized for prot
  float* tb   = ws + off; off += (size_t)NPROT * HDIM;
  float* v_l  = ws + off; off += (size_t)BGRAPH * HDIM;
  float* v_p  = ws + off; off += (size_t)BGRAPH * HDIM;
  float* vt   = ws + off; off += (size_t)BGRAPH * HDIM;
  float* mean = ws + off; off += HDIM;
  float* var  = ws + off; off += HDIM;
  float* comb = ws + off; off += (size_t)BGRAPH * 2 * HDIM;
  float* h1   = ws + off; off += (size_t)BGRAPH * HDIM;
  unsigned short* wp = (unsigned short*)(ws + off);      // up to 512*256 bf16 = 256 KB

  // ---- embeddings + virtual-node init ----
  embed_kernel<<<NLIG, 256, 0, stream>>>(lig_x,  lig_emb_W,  lig_emb_b,  h_l, 26);
  embed_kernel<<<NPROT, 256, 0, stream>>>(prot_x, prot_emb_W, prot_emb_b, h_p, 20);
  bcast_init<<<BGRAPH * HDIM / 256, 256, 0, stream>>>(lig_v0,  v_l);
  bcast_init<<<BGRAPH * HDIM / 256, 256, 0, stream>>>(prot_v0, v_p);

  const size_t HH = (size_t)HDIM * HDIM;
  for (int li = 0; li < LAYERS; ++li) {
    // ---------------- ligand branch ----------------
    pool_add<<<BGRAPH, 256, 0, stream>>>(h_l, v_l, NLIG / BGRAPH);
    pack_w<<<(HDIM / 32) * (HDIM / 16), 32, 0, stream>>>(lig_vW + li * HH, wp);
    gemm256_wmma<<<BGRAPH / 32, 256, 0, stream>>>(v_l, nullptr, HDIM, wp,
                                                  lig_vb + li * HDIM, nullptr, vt, HDIM, 0);
    colstats<<<HDIM, 256, 0, stream>>>(vt, mean, var);
    bn_relu_apply<<<BGRAPH * HDIM / 256, 256, 0, stream>>>(vt, mean, var,
                                                           lig_vg + li * HDIM,
                                                           lig_vbeta + li * HDIM, v_l);
    vadd_bcast<<<NLIG * HDIM / 256, 256, 0, stream>>>(h_l, v_l, 6);
    zero_kernel<<<NLIG * HDIM / 256, 256, 0, stream>>>(agg, (size_t)NLIG * HDIM);
    edge_scatter<<<ELIG * 32 / 256, 256, 0, stream>>>(h_l, lig_edge, lig_edge + ELIG, agg, ELIG);
    pack_w<<<(HDIM / 32) * (HDIM / 16), 32, 0, stream>>>(lig_cW1 + li * HH, wp);
    gemm256_wmma<<<NLIG / 32, 256, 0, stream>>>(h_l, agg, HDIM, wp,
                                                lig_cb1 + li * HDIM, nullptr, tb, HDIM, 1);
    pack_w<<<(HDIM / 32) * (HDIM / 16), 32, 0, stream>>>(lig_cW2 + li * HH, wp);
    gemm256_wmma<<<NLIG / 32, 256, 0, stream>>>(tb, nullptr, HDIM, wp,
                                                lig_cb2 + li * HDIM, h_l, h_l, HDIM, 0);

    // ---------------- protein branch ----------------
    pool_add<<<BGRAPH, 256, 0, stream>>>(h_p, v_p, NPROT / BGRAPH);
    pack_w<<<(HDIM / 32) * (HDIM / 16), 32, 0, stream>>>(prot_vW + li * HH, wp);
    gemm256_wmma<<<BGRAPH / 32, 256, 0, stream>>>(v_p, nullptr, HDIM, wp,
                                                  prot_vb + li * HDIM, nullptr, vt, HDIM, 0);
    colstats<<<HDIM, 256, 0, stream>>>(vt, mean, var);
    bn_relu_apply<<<BGRAPH * HDIM / 256, 256, 0, stream>>>(vt, mean, var,
                                                           prot_vg + li * HDIM,
                                                           prot_vbeta + li * HDIM, v_p);
    vadd_bcast<<<NPROT * HDIM / 256, 256, 0, stream>>>(h_p, v_p, 7);
    zero_kernel<<<NPROT * HDIM / 256, 256, 0, stream>>>(agg, (size_t)NPROT * HDIM);
    edge_scatter<<<EPROT * 32 / 256, 256, 0, stream>>>(h_p, prot_edge, prot_edge + EPROT, agg, EPROT);
    pack_w<<<(HDIM / 32) * (HDIM / 16), 32, 0, stream>>>(prot_cW1 + li * HH, wp);
    gemm256_wmma<<<NPROT / 32, 256, 0, stream>>>(h_p, agg, HDIM, wp,
                                                 prot_cb1 + li * HDIM, nullptr, tb, HDIM, 1);
    pack_w<<<(HDIM / 32) * (HDIM / 16), 32, 0, stream>>>(prot_cW2 + li * HH, wp);
    gemm256_wmma<<<NPROT / 32, 256, 0, stream>>>(tb, nullptr, HDIM, wp,
                                                 prot_cb2 + li * HDIM, h_p, h_p, HDIM, 0);
  }

  // ---- readout + prediction MLP ----
  pool_mean<<<BGRAPH, 256, 0, stream>>>(h_l, comb, NLIG / BGRAPH, 0);
  pool_mean<<<BGRAPH, 256, 0, stream>>>(h_p, comb, NPROT / BGRAPH, HDIM);
  pack_w<<<(2 * HDIM / 32) * (HDIM / 16), 32, 0, stream>>>(pred_W1, wp);  // K = 512
  gemm256_wmma<<<BGRAPH / 32, 256, 0, stream>>>(comb, nullptr, 2 * HDIM, wp,
                                                pred_b1, nullptr, h1, 2 * HDIM, 1);
  final_dot<<<BGRAPH / 8, 256, 0, stream>>>(h1, pred_W2, pred_b2, (float*)d_out);
}
